// GLMAttention_38517266710910
// MI455X (gfx1250) — matrix operand order repaired and checked by reading, not compiled, not run on hardware
//
#include <hip/hip_runtime.h>
#include <hip/hip_bf16.h>

// ---------------- problem constants ----------------
#define BB    2
#define SS    1024
#define HH    4096
#define NHD   32
#define DHD   128
#define NKVH  2
#define GRP   16        // NH / NKV
#define NQ    (NHD*DHD)           // 4096
#define NQKV  (NQ + 2*NKVH*DHD)   // 4608
#define MTOT  (BB*SS)             // 2048
#define QSCALE 0.08838834764831845f   // 1/sqrt(128)

// ---------------- WMMA / TDM types ----------------
typedef __attribute__((ext_vector_type(16))) __bf16 v16bf;
typedef __attribute__((ext_vector_type(8)))  float  v8f;
typedef __attribute__((ext_vector_type(4)))  unsigned int u32x4;
typedef __attribute__((ext_vector_type(8)))  int i32x8;
typedef __attribute__((ext_vector_type(4)))  int i32x4;

union Frag { v16bf f; uint4 u[2]; };

static __device__ inline v8f vzero8() {
  v8f z = {0.f,0.f,0.f,0.f,0.f,0.f,0.f,0.f};
  return z;
}

static __device__ inline v8f wmma_bf16(v16bf a, v16bf b, v8f c) {
  // (neg_a, A, neg_b, B, c_mod, C, reuse_a, reuse_b)
  return __builtin_amdgcn_wmma_f32_16x16x32_bf16(false, a, false, b, (short)0, c, false, false);
}

static __device__ inline unsigned short f32_bf16(float f) {
  unsigned int u = __float_as_uint(f);
  u += 0x7FFFu + ((u >> 16) & 1u);   // round-to-nearest-even
  return (unsigned short)(u >> 16);
}

// ---- TDM: issue a 2D tile load (tile = 64 rows x 32 bf16, row-major into LDS) ----
// D# group0: count=1, lds_addr, global tile addr, type=2 (image)
// D# group1: data_size=2B, tensor_dim0=Kd, tensor_dim1=Mrows, tile 32x64, dim0_stride=Kd
static __device__ inline void tdm_load_tile64x32(unsigned lds_off,
                                                 const unsigned short* gptr,
                                                 int Kd, int Mrows) {
  unsigned long long ga = (unsigned long long)(size_t)gptr;
  u32x4 g0;
  g0[0] = 1u;                                        // count=1 (valid), user mode
  g0[1] = lds_off;                                   // lds_addr (bytes)
  g0[2] = (unsigned)(ga & 0xFFFFFFFFu);              // global_addr[31:0]
  g0[3] = (unsigned)((ga >> 32) & 0x01FFFFFFu)       // global_addr[56:32]
        | (2u << 30);                                // type = 2
  i32x8 g1;
  g1[0] = (int)(1u << 16);                           // workgroup_mask=0, data_size=1 (2B)
  g1[1] = (int)(((unsigned)Kd & 0xFFFFu) << 16);     // tensor_dim0[15:0] @ bits[31:16]
  g1[2] = (int)(((unsigned)Kd >> 16)                 // tensor_dim0[31:16]
        | (((unsigned)Mrows & 0xFFFFu) << 16));      // tensor_dim1[15:0]
  g1[3] = (int)((((unsigned)Mrows >> 16) & 0xFFFFu)  // tensor_dim1[31:16]
        | (32u << 16));                              // tile_dim0 = 32 elements
  g1[4] = (int)64u;                                  // tile_dim1 = 64, tile_dim2 = 0
  g1[5] = (int)Kd;                                   // tensor_dim0_stride[31:0]
  g1[6] = 0;                                         // stride[47:32]=0, dim1_stride lo=0
  g1[7] = 0;
  i32x4 z4 = {0, 0, 0, 0};
#if defined(__clang_major__) && (__clang_major__ >= 23)
  i32x8 z8 = {0, 0, 0, 0, 0, 0, 0, 0};
  __builtin_amdgcn_tensor_load_to_lds(g0, g1, z4, z4, z8, 0);
#else
  __builtin_amdgcn_tensor_load_to_lds(g0, g1, z4, z4, 0);
#endif
}

// ---------------- 0: fp32 -> bf16 row-major ----------------
__global__ void cvt_bf16_kernel(const float* __restrict__ in,
                                unsigned short* __restrict__ out, int n) {
  int i = blockIdx.x * blockDim.x + threadIdx.x;
  if (i < n) out[i] = f32_bf16(in[i]);
}

// ---------------- 1: pack W[K,N] fp32 into WMMA B-fragment bf16 layout ------
// Fragment (kc, nt) = 32 lanes x 16 bf16.  For lane l (half = l>>4, n = nt*16 + (l&15)),
// element e (0..15) holds  W[kc*32 + half*16 + e][n].
__global__ void pack_w_kernel(const float* __restrict__ W,
                              unsigned short* __restrict__ out, int Kd, int N) {
  int i = blockIdx.x * blockDim.x + threadIdx.x;
  int total = Kd * N;
  if (i >= total) return;
  int nk   = Kd >> 5;
  int e    = i & 15;
  int lane = (i >> 4) & 31;
  int kc   = (i >> 9) % nk;
  int nt   = i / (nk << 9);
  int half = lane >> 4;
  int k    = kc * 32 + half * 16 + e;
  int n    = nt * 16 + (lane & 15);
  out[i] = f32_bf16(W[(size_t)k * N + n]);
}

// ---------------- 2: bf16 GEMM  C[M,N] = A[M,Kd] * Bpacked ----------------
// 256 threads = 8 waves; WG tile 64(M) x 128(N); wave tile 16x64 = 4 WMMAs per
// K-step reusing one A fragment.  A tile (64x32 bf16 = 4KB) staged into LDS by
// the Tensor Data Mover, double-buffered, gated by s_wait_tensorcnt + barrier.
__global__ __launch_bounds__(256) void gemm_bf16_kernel(
    const unsigned short* __restrict__ A,
    const unsigned short* __restrict__ Bp,
    float* __restrict__ C, int M, int N, int Kd) {
  __shared__ __align__(16) unsigned short As[2][64 * 32];

  int tid  = threadIdx.x;
  int wv   = tid >> 5;
  int l    = tid & 31;
  int half = l >> 4;
  int ln   = l & 15;
  int wy   = wv >> 1;          // 0..3 (M sub-tile)
  int wx   = wv & 1;           // 0..1 (N quad pair)
  int m0   = blockIdx.x * 64;
  int n0   = blockIdx.y * 128;
  int nk   = Kd >> 5;

  unsigned ldsbase = (unsigned)(size_t)(&As[0][0]);
  const unsigned short* aTile = A + (size_t)m0 * Kd;

  // prime buffer 0 via TDM (wave 0 issues; TENSORcnt tracked per issuing wave)
  if (wv == 0) {
    tdm_load_tile64x32(ldsbase, aTile, Kd, M);
    __builtin_amdgcn_s_wait_tensorcnt(0);
  }
  __syncthreads();

  v8f acc[4];
  for (int j = 0; j < 4; ++j) acc[j] = vzero8();

  int arow = wy * 16 + ln;
  int ntg  = blockIdx.y * 8 + wx * 4;
  const unsigned short* bbase[4];
  for (int j = 0; j < 4; ++j)
    bbase[j] = Bp + ((size_t)(ntg + j) * nk) * 512 + l * 16;

  for (int kc = 0; kc < nk; ++kc) {
    int cur = kc & 1;
    // kick off next tile DMA while this one is consumed
    if (wv == 0 && kc + 1 < nk)
      tdm_load_tile64x32(ldsbase + 4096u * ((kc + 1) & 1),
                         aTile + (size_t)(kc + 1) * 32, Kd, M);

    Frag a;
    a.u[0] = *(const uint4*)(&As[cur][arow * 32 + half * 16]);
    a.u[1] = *(const uint4*)(&As[cur][arow * 32 + half * 16 + 8]);

    if (kc + 4 < nk) {
      __builtin_prefetch(bbase[0] + (size_t)(kc + 4) * 512, 0, 1);
      __builtin_prefetch(bbase[2] + (size_t)(kc + 4) * 512, 0, 1);
    }
#pragma unroll
    for (int j = 0; j < 4; ++j) {
      const unsigned short* b = bbase[j] + (size_t)kc * 512;
      Frag fb;
      fb.u[0] = *(const uint4*)(b);
      fb.u[1] = *(const uint4*)(b + 8);
      acc[j] = wmma_bf16(a.f, fb.f, acc[j]);
    }

    // next buffer must be resident before anyone enters the next iteration
    if (wv == 0 && kc + 1 < nk)
      __builtin_amdgcn_s_wait_tensorcnt(0);
    __syncthreads();
  }

  int c0 = n0 + wx * 64 + ln;
  for (int r = 0; r < 8; ++r) {
    int rr = m0 + wy * 16 + r + 8 * half;
    float* crow = C + (size_t)rr * N + c0;
#pragma unroll
    for (int j = 0; j < 4; ++j) crow[j * 16] = acc[j][r];
  }
}

// ---------------- 3: bias + GLM interleaved RoPE + pack to bf16 -------------
static __device__ inline void rope2(float& x0, float& x1, int d, int pos) {
  // pair (d, d+1), d even, d < 64 ; inv_freq = 10000^(-d/64)
  float inv = __powf(10000.0f, -(float)d * (1.0f / 64.0f));
  float ang = (float)pos * inv;
  float cs = __cosf(ang), sn = __sinf(ang);
  float o0 = x0 * cs - x1 * sn;
  float o1 = x1 * cs + x0 * sn;
  x0 = o0; x1 = o1;
}

__global__ void rope_pack_kernel(const float* __restrict__ qkv,
                                 const float* __restrict__ bias,
                                 const int* __restrict__ pos_ids,
                                 unsigned short* __restrict__ Qb,   // [B,NH,S,DH]
                                 unsigned short* __restrict__ Kb,   // [B,NKV,S,DH]
                                 unsigned short* __restrict__ Vt) { // [B,NKV,DH,S]
  int i = blockIdx.x * blockDim.x + threadIdx.x;
  const int PAIRS = NQKV / 2;                  // 2304
  if (i >= MTOT * PAIRS) return;
  int t = i / PAIRS;
  int c = (i % PAIRS) * 2;
  int b = t >> 10;
  int s = t & (SS - 1);
  float x0 = qkv[(size_t)t * NQKV + c]     + bias[c];
  float x1 = qkv[(size_t)t * NQKV + c + 1] + bias[c + 1];
  int pos = pos_ids[t];

  if (c < NQ) {                                // Q
    int h = c >> 7, d = c & 127;
    if (d < 64) rope2(x0, x1, d, pos);
    size_t o = (((size_t)b * NHD + h) * SS + s) * DHD + d;
    Qb[o]     = f32_bf16(x0 * QSCALE);
    Qb[o + 1] = f32_bf16(x1 * QSCALE);
  } else if (c < NQ + NKVH * DHD) {            // K
    int cc = c - NQ;
    int kh = cc >> 7, d = cc & 127;
    if (d < 64) rope2(x0, x1, d, pos);
    size_t o = (((size_t)b * NKVH + kh) * SS + s) * DHD + d;
    Kb[o]     = f32_bf16(x0);
    Kb[o + 1] = f32_bf16(x1);
  } else {                                     // V (store transposed)
    int cc = c - NQ - NKVH * DHD;
    int kh = cc >> 7, d = cc & 127;
    size_t o = (((size_t)b * NKVH + kh) * DHD + d) * SS + s;
    Vt[o]      = f32_bf16(x0);
    Vt[o + SS] = f32_bf16(x1);
  }
}

// ---------------- 4: fused causal flash attention (bf16 WMMA) --------------
// block = 128 threads (4 waves); wave -> 16 q rows; k-tiles of 32.
__global__ __launch_bounds__(128) void attn_kernel(
    const unsigned short* __restrict__ Qb,
    const unsigned short* __restrict__ Kb,
    const unsigned short* __restrict__ Vt,
    unsigned short* __restrict__ ctx) {       // [B*S, NH*DH] bf16
  __shared__ __align__(16) unsigned short pls[4][16 * 32];

  int tid  = threadIdx.x;
  int wv   = tid >> 5;
  int l    = tid & 31;
  int half = l >> 4;
  int ln   = l & 15;

  int idx  = blockIdx.x;                  // b*512 + h*16 + qb
  int b    = idx >> 9;
  int h    = (idx >> 4) & 31;
  int q0w  = (idx & 15) * 64 + wv * 16;
  int kh   = h >> 4;                      // GQA group of 16

  // preload Q fragments (16 rows x 128 dh = 4 A-fragments)
  Frag qf[4];
  const unsigned short* qp = Qb + (((size_t)b * NHD + h) * SS + (q0w + ln)) * DHD;
  for (int cf = 0; cf < 4; ++cf) {
    qf[cf].u[0] = *(const uint4*)(qp + cf * 32 + half * 16);
    qf[cf].u[1] = *(const uint4*)(qp + cf * 32 + half * 16 + 8);
  }

  v8f o[8];
  for (int nt = 0; nt < 8; ++nt) o[nt] = vzero8();
  float mrow[8], lrow[8];
  for (int r = 0; r < 8; ++r) { mrow[r] = -3.0e38f; lrow[r] = 0.f; }

  const unsigned short* Kbase = Kb + ((size_t)b * NKVH + kh) * SS * DHD;
  const unsigned short* Vbase = Vt + ((size_t)b * NKVH + kh) * DHD * SS;
  unsigned short* pl = &pls[wv][0];

  int nkt = ((q0w + 15) >> 5) + 1;
  for (int kt = 0; kt < nkt; ++kt) {
    int kbase = kt * 32;

    // ---- scores: 16 q x 32 k ----
    v8f s0 = vzero8(), s1 = vzero8();
    const unsigned short* k0p = Kbase + (size_t)(kbase + ln) * DHD;
    const unsigned short* k1p = Kbase + (size_t)(kbase + 16 + ln) * DHD;
    for (int cf = 0; cf < 4; ++cf) {
      Frag kf0, kf1;
      kf0.u[0] = *(const uint4*)(k0p + cf * 32 + half * 16);
      kf0.u[1] = *(const uint4*)(k0p + cf * 32 + half * 16 + 8);
      kf1.u[0] = *(const uint4*)(k1p + cf * 32 + half * 16);
      kf1.u[1] = *(const uint4*)(k1p + cf * 32 + half * 16 + 8);
      s0 = wmma_bf16(qf[cf].f, kf0.f, s0);
      s1 = wmma_bf16(qf[cf].f, kf1.f, s1);
    }

    // ---- masked online softmax (per output row r) ----
    float a8[8];
    for (int r = 0; r < 8; ++r) {
      int qrow = q0w + r + 8 * half;
      float v0 = (kbase + ln      <= qrow) ? s0[r] : -3.0e38f;
      float v1 = (kbase + 16 + ln <= qrow) ? s1[r] : -3.0e38f;
      float rmax = fmaxf(v0, v1);
      for (int mm = 8; mm; mm >>= 1) rmax = fmaxf(rmax, __shfl_xor(rmax, mm, 16));
      float mnew = fmaxf(mrow[r], rmax);
      float p0 = __expf(v0 - mnew);
      float p1 = __expf(v1 - mnew);
      float ps = p0 + p1;
      for (int mm = 8; mm; mm >>= 1) ps += __shfl_xor(ps, mm, 16);
      float alpha = __expf(mrow[r] - mnew);
      mrow[r] = mnew;
      lrow[r] = lrow[r] * alpha + ps;
      a8[r] = alpha;
      int rr = r + 8 * half;
      pl[rr * 32 + ln]      = f32_bf16(p0);
      pl[rr * 32 + 16 + ln] = f32_bf16(p1);
    }
    for (int nt = 0; nt < 8; ++nt)
      for (int r = 0; r < 8; ++r) o[nt][r] *= a8[r];

    // reload P as A-fragment (per-wave LDS region; wave-internal ordering)
    Frag pf;
    pf.u[0] = *(const uint4*)(pl + ln * 32 + half * 16);
    pf.u[1] = *(const uint4*)(pl + ln * 32 + half * 16 + 8);

    // ---- ctx += P (16x32) * V (32x128) ----
    for (int nt = 0; nt < 8; ++nt) {
      const unsigned short* vp = Vbase + (size_t)(nt * 16 + ln) * SS + kbase + half * 16;
      Frag vf;
      vf.u[0] = *(const uint4*)(vp);
      vf.u[1] = *(const uint4*)(vp + 8);
      o[nt] = wmma_bf16(pf.f, vf.f, o[nt]);
    }
  }

  // ---- normalize + store bf16 ctx ----
  for (int r = 0; r < 8; ++r) {
    int qrow = q0w + r + 8 * half;
    float inv = 1.0f / lrow[r];
    size_t ro = ((size_t)b * SS + qrow) * NQ + (size_t)h * DHD + ln;
    for (int nt = 0; nt < 8; ++nt)
      ctx[ro + nt * 16] = f32_bf16(o[nt][r] * inv);
  }
}

// ---------------- launcher ----------------
extern "C" void kernel_launch(void* const* d_in, const int* in_sizes, int n_in,
                              void* d_out, int out_size, void* d_ws, size_t ws_size,
                              hipStream_t stream) {
  const float* hidden  = (const float*)d_in[0];
  const int*   pos_ids = (const int*)  d_in[1];
  const float* W_qkv   = (const float*)d_in[2];
  const float* b_qkv   = (const float*)d_in[3];
  const float* W_dense = (const float*)d_in[4];
  float* out = (float*)d_out;
  (void)in_sizes; (void)n_in; (void)out_size; (void)ws_size;

  size_t off = 0;
  auto alloc = [&](size_t bytes) -> void* {
    void* p = (char*)d_ws + off;
    off += (bytes + 255) & ~(size_t)255;
    return p;
  };
  unsigned short* hb    = (unsigned short*)alloc((size_t)MTOT * HH * 2);       // hidden bf16
  unsigned short* wqkvp = (unsigned short*)alloc((size_t)HH * NQKV * 2);       // packed W_qkv
  unsigned short* wdp   = (unsigned short*)alloc((size_t)NQ * HH * 2);         // packed W_dense
  float*          qkvf  = (float*)        alloc((size_t)MTOT * NQKV * 4);      // qkv fp32
  unsigned short* Qb    = (unsigned short*)alloc((size_t)BB * NHD * SS * DHD * 2);
  unsigned short* Kb    = (unsigned short*)alloc((size_t)BB * NKVH * SS * DHD * 2);
  unsigned short* Vtb   = (unsigned short*)alloc((size_t)BB * NKVH * DHD * SS * 2);
  unsigned short* ctxb  = (unsigned short*)alloc((size_t)MTOT * NQ * 2);

  // 0) activations fp32 -> bf16
  {
    int n = MTOT * HH;
    cvt_bf16_kernel<<<(n + 255) / 256, 256, 0, stream>>>(hidden, hb, n);
  }
  // 1) pack weights into WMMA B-fragment layout
  {
    int n = HH * NQKV;
    pack_w_kernel<<<(n + 255) / 256, 256, 0, stream>>>(W_qkv, wqkvp, HH, NQKV);
  }
  {
    int n = NQ * HH;
    pack_w_kernel<<<(n + 255) / 256, 256, 0, stream>>>(W_dense, wdp, NQ, HH);
  }
  // 2) QKV GEMM  [2048,4096] x [4096,4608] -> fp32 (TDM-staged A tiles)
  {
    dim3 grid(MTOT / 64, NQKV / 128);
    gemm_bf16_kernel<<<grid, 256, 0, stream>>>(hb, wqkvp, qkvf, MTOT, NQKV, HH);
  }
  // 3) bias + RoPE + pack to bf16 Q/K/V^T (Q pre-scaled)
  {
    int n = MTOT * (NQKV / 2);
    rope_pack_kernel<<<(n + 255) / 256, 256, 0, stream>>>(qkvf, b_qkv, pos_ids, Qb, Kb, Vtb);
  }
  // 4) fused causal flash attention
  {
    attn_kernel<<<BB * NHD * (SS / 64), 128, 0, stream>>>(Qb, Kb, Vtb, ctxb);
  }
  // 5) dense projection  [2048,4096] x [4096,4096] -> out fp32 (TDM-staged)
  {
    dim3 grid(MTOT / 64, HH / 128);
    gemm_bf16_kernel<<<grid, 256, 0, stream>>>(ctxb, wdp, out, MTOT, HH, HH);
  }
}
